// snnLlamaMLP_80058190397925
// MI455X (gfx1250) — matrix operand order repaired and checked by reading, not compiled, not run on hardware
//
#include <hip/hip_runtime.h>
#include <hip/hip_bf16.h>

// ---- types ----------------------------------------------------------------
typedef __attribute__((ext_vector_type(16))) __bf16 bf16x16;
typedef __attribute__((ext_vector_type(8)))  __bf16 bf16x8;
typedef __attribute__((ext_vector_type(4)))  __bf16 bf16x4;
typedef __attribute__((ext_vector_type(8)))  float  v8f;
typedef __attribute__((ext_vector_type(4)))  float  v4f;

union FragBF { bf16x16 v; bf16x8 h[2]; };

static constexpr int TSTEPS = 4;
static constexpr int HID    = 2048;
static constexpr int INTER  = 5504;
static constexpr int BSROWS = 1024;            // bsz * seq
static constexpr int MTOT   = TSTEPS * BSROWS; // 4096

// ---- f32 -> bf16 conversion (4 elems / thread) ----------------------------
__global__ void cvt_f32_to_bf16_kernel(const float* __restrict__ src,
                                       __bf16* __restrict__ dst, int n) {
  int i = (blockIdx.x * blockDim.x + threadIdx.x) * 4;
  if (i + 4 <= n) {
    v4f f = *reinterpret_cast<const v4f*>(src + i);
    bf16x4 b;
    b.x = (__bf16)f.x; b.y = (__bf16)f.y; b.z = (__bf16)f.z; b.w = (__bf16)f.w;
    *reinterpret_cast<bf16x4*>(dst + i) = b;
  }
}

// ---- fused gate/up GEMM + temporal SiLU/hadamard epilogue -----------------
// Block: 256 thr = 8 waves (4 M-waves x 2 N-waves). Block tile: 64 rows of
// (b,s)-space x 64 inter-cols, for ALL 4 timesteps. Per wave: 16x32 tile per
// timestep per weight -> 16 accumulators (128 VGPRs), 16 WMMAs per K-step.
__global__ __launch_bounds__(256, 1)
void gateup_act_kernel(const __bf16* __restrict__ xb,
                       const __bf16* __restrict__ wg,
                       const __bf16* __restrict__ wu,
                       __bf16* __restrict__ cb) {
  const int lane = threadIdx.x & 31;
  const int wave = threadIdx.x >> 5;
  const int wm   = wave & 3;           // 0..3 : 16-row slice
  const int wn   = wave >> 2;          // 0..1 : 32-col slice
  const int half = lane >> 4;          // wave32 half
  const int lm   = lane & 15;

  const int mBase = blockIdx.x * 64 + wm * 16;   // row in (b,s)-space [0,1024)
  const int nBase = blockIdx.y * 64 + wn * 32;   // col in inter space

  // A rows: one per timestep (global row = t*1024 + mBase + lm)
  const __bf16* aP[TSTEPS];
#pragma unroll
  for (int t = 0; t < TSTEPS; ++t)
    aP[t] = xb + (size_t)(t * BSROWS + mBase + lm) * HID + half * 8;

  // B rows (weights are (N,K) row-major -> K-contiguous columns of B)
  const __bf16* bgP[2];
  const __bf16* buP[2];
#pragma unroll
  for (int j = 0; j < 2; ++j) {
    size_t r = (size_t)(nBase + j * 16 + lm) * HID + half * 8;
    bgP[j] = wg + r;
    buP[j] = wu + r;
  }

  v8f accG[TSTEPS][2];
  v8f accU[TSTEPS][2];
#pragma unroll
  for (int t = 0; t < TSTEPS; ++t)
#pragma unroll
    for (int j = 0; j < 2; ++j) { accG[t][j] = {}; accU[t][j] = {}; }

  for (int k = 0; k < HID; k += 32) {
    FragBF bg[2], bu[2], a[TSTEPS];
#pragma unroll
    for (int j = 0; j < 2; ++j) {
      bg[j].h[0] = *reinterpret_cast<const bf16x8*>(bgP[j] + k);
      bg[j].h[1] = *reinterpret_cast<const bf16x8*>(bgP[j] + k + 16);
      bu[j].h[0] = *reinterpret_cast<const bf16x8*>(buP[j] + k);
      bu[j].h[1] = *reinterpret_cast<const bf16x8*>(buP[j] + k + 16);
    }
#pragma unroll
    for (int t = 0; t < TSTEPS; ++t) {
      a[t].h[0] = *reinterpret_cast<const bf16x8*>(aP[t] + k);
      a[t].h[1] = *reinterpret_cast<const bf16x8*>(aP[t] + k + 16);
    }
    if (k + 256 < HID) {   // prefetch streamed activations (global_prefetch_b8)
#pragma unroll
      for (int t = 0; t < TSTEPS; ++t) __builtin_prefetch(aP[t] + k + 256, 0, 1);
    }
#pragma unroll
    for (int t = 0; t < TSTEPS; ++t)
#pragma unroll
      for (int j = 0; j < 2; ++j) {
        accG[t][j] = __builtin_amdgcn_wmma_f32_16x16x32_bf16(
            false, a[t].v, false, bg[j].v, (short)0, accG[t][j], false, false);
        accU[t][j] = __builtin_amdgcn_wmma_f32_16x16x32_bf16(
            false, a[t].v, false, bu[j].v, (short)0, accU[t][j], false, false);
      }
  }

  // In-register temporal epilogue:
  //   X_t = cumsum(gate), Y = silu(X), A_t = Y_t - Y_{t-1}
  //   C_t = 0.5 * (A_t * sum(up) + up_t * sum(A))
#pragma unroll
  for (int j = 0; j < 2; ++j) {
    const int col = nBase + j * 16 + lm;
#pragma unroll
    for (int e = 0; e < 8; ++e) {
      const int row = mBase + e + half * 8;   // C/D frag: M = e + 8*half
      float X = 0.f, Yp = 0.f, SA = 0.f, SB = 0.f;
      float A[TSTEPS], U[TSTEPS];
#pragma unroll
      for (int t = 0; t < TSTEPS; ++t) {
        X += accG[t][j][e];
        float Y = X * (1.f / (1.f + __expf(-X)));   // silu
        A[t] = Y - Yp; Yp = Y; SA += A[t];
        U[t] = accU[t][j][e]; SB += U[t];
      }
#pragma unroll
      for (int t = 0; t < TSTEPS; ++t) {
        float c = 0.5f * (A[t] * SB + U[t] * SA);
        cb[(size_t)(t * BSROWS + row) * INTER + col] = (__bf16)c;
      }
    }
  }
}

// ---- down projection GEMM with LDS-staged B -------------------------------
// Block: 256 thr = 8 waves stacked in M. Block tile 256x64; wave tile 32x64
// (8 accumulators, 8 WMMAs per K-step). w_down tile (64 n-rows x 32 k) is
// staged once per block into double-buffered LDS (ds_load/ds_store path),
// removing the 8x redundant per-wave global B traffic. LDS rows padded to
// 40 elems (80 B = 20 banks, gcd(5,16)=1) -> conflict-free b128 frag reads.
__global__ __launch_bounds__(256, 1)
void down_kernel(const __bf16* __restrict__ cb,
                 const __bf16* __restrict__ wd,
                 float* __restrict__ out) {
  constexpr int LROW  = 40;             // padded LDS row stride (elems)
  constexpr int NSTEP = INTER / 32;     // 172
  __shared__ __bf16 lbs[2][64 * LROW];  // 10 KB

  const int lane = threadIdx.x & 31;
  const int wave = threadIdx.x >> 5;
  const int half = lane >> 4;
  const int lm   = lane & 15;

  const int mBase = blockIdx.x * 256 + wave * 32;  // rows in [0,4096)
  const int nBase = blockIdx.y * 64;               // cols in hidden space

  // A pointers: two 16-row groups per wave
  const __bf16* aP[2];
#pragma unroll
  for (int r = 0; r < 2; ++r)
    aP[r] = cb + (size_t)(mBase + r * 16 + lm) * INTER + half * 8;

  // cooperative B staging: 64 rows x 32 k = 256 x b128 chunks, one per thread
  const int srow = threadIdx.x >> 2;    // 0..63
  const int sseg = threadIdx.x & 3;     // 0..3 (8-elem segment)
  const __bf16* gsrc = wd + (size_t)(nBase + srow) * INTER + sseg * 8;
  const int loff = srow * LROW + sseg * 8;

  v8f acc[2][4];
#pragma unroll
  for (int r = 0; r < 2; ++r)
#pragma unroll
    for (int j = 0; j < 4; ++j) acc[r][j] = {};

  // prologue: stage tile 0
  {
    bf16x8 s0 = *reinterpret_cast<const bf16x8*>(gsrc);
    *reinterpret_cast<bf16x8*>(&lbs[0][loff]) = s0;
  }

  for (int step = 0; step < NSTEP; ++step) {
    __syncthreads();                    // staged stores of buf[step&1] visible
    const int k = step * 32;
    const bool more = (step + 1 < NSTEP);

    bf16x8 nxt = {};
    if (more) nxt = *reinterpret_cast<const bf16x8*>(gsrc + (step + 1) * 32);

    FragBF b[4];
#pragma unroll
    for (int j = 0; j < 4; ++j) {
      const __bf16* lp = &lbs[step & 1][(j * 16 + lm) * LROW + half * 8];
      b[j].h[0] = *reinterpret_cast<const bf16x8*>(lp);
      b[j].h[1] = *reinterpret_cast<const bf16x8*>(lp + 16);
    }

    FragBF a[2];
#pragma unroll
    for (int r = 0; r < 2; ++r) {
      a[r].h[0] = *reinterpret_cast<const bf16x8*>(aP[r] + k);
      a[r].h[1] = *reinterpret_cast<const bf16x8*>(aP[r] + k + 16);
    }
    if (k + 512 < INTER) {              // prefetch streamed activations
#pragma unroll
      for (int r = 0; r < 2; ++r) __builtin_prefetch(aP[r] + k + 512, 0, 1);
    }

#pragma unroll
    for (int r = 0; r < 2; ++r)
#pragma unroll
      for (int j = 0; j < 4; ++j)
        acc[r][j] = __builtin_amdgcn_wmma_f32_16x16x32_bf16(
            false, a[r].v, false, b[j].v, (short)0, acc[r][j], false, false);

    if (more)                           // safe: readers of buf[(step+1)&1]
      *reinterpret_cast<bf16x8*>(&lbs[(step + 1) & 1][loff]) = nxt;
  }

#pragma unroll
  for (int r = 0; r < 2; ++r)
#pragma unroll
    for (int j = 0; j < 4; ++j) {
      const int col = nBase + j * 16 + lm;
#pragma unroll
      for (int e = 0; e < 8; ++e) {
        const int row = mBase + r * 16 + e + half * 8;
        out[(size_t)row * HID + col] = acc[r][j][e];
      }
    }
}

// ---- launcher -------------------------------------------------------------
extern "C" void kernel_launch(void* const* d_in, const int* in_sizes, int n_in,
                              void* d_out, int out_size, void* d_ws, size_t ws_size,
                              hipStream_t stream) {
  const float* x  = (const float*)d_in[0];   // (8, 512, 2048)
  const float* wg = (const float*)d_in[1];   // (5504, 2048)
  const float* wu = (const float*)d_in[2];   // (5504, 2048)
  const float* wd = (const float*)d_in[3];   // (2048, 5504)

  const size_t NX = (size_t)MTOT * HID;      // 8,388,608
  const size_t NW = (size_t)INTER * HID;     // 11,272,192
  const size_t NC = (size_t)MTOT * INTER;    // 22,544,384

  char* p = (char*)d_ws;
  auto alloc = [&](size_t bytes) {
    char* r = p;
    p += (bytes + 255) & ~(size_t)255;
    return r;
  };
  __bf16* xb  = (__bf16*)alloc(NX * 2);
  __bf16* wgb = (__bf16*)alloc(NW * 2);
  __bf16* wub = (__bf16*)alloc(NW * 2);
  __bf16* wdb = (__bf16*)alloc(NW * 2);
  __bf16* cbf = (__bf16*)alloc(NC * 2);

  auto cvt = [&](const float* s, __bf16* d, size_t n) {
    int blocks = (int)((n / 4 + 255) / 256);
    cvt_f32_to_bf16_kernel<<<blocks, 256, 0, stream>>>(s, d, (int)n);
  };
  cvt(x,  xb,  NX);
  cvt(wg, wgb, NW);
  cvt(wu, wub, NW);
  cvt(wd, wdb, NW);

  dim3 g1(BSROWS / 64, INTER / 64);   // (16, 86)
  gateup_act_kernel<<<g1, 256, 0, stream>>>(xb, wgb, wub, cbf);

  dim3 g2(MTOT / 256, HID / 64);      // (16, 32)
  down_kernel<<<g2, 256, 0, stream>>>(cbf, wdb, (float*)d_out);
}
